// MultiHeadSelfAttention_60808146977175
// MI455X (gfx1250) — compile-verified, matrix operand order checked by reading, not run on hardware
//
#include <hip/hip_runtime.h>
#include <cstdint>
#include <cstddef>

// Problem constants (match reference)
#define B_   2
#define T_   2048
#define C_   1024
#define H_   16
#define DH_  64
#define M_   (B_ * T_)      // 4096 tokens
#define K_   C_             // GEMM reduction dim
#define N_   C_             // GEMM output dim

typedef __attribute__((ext_vector_type(16))) __bf16 v16bf;
typedef __attribute__((ext_vector_type(8)))  float  v8f;
typedef __attribute__((ext_vector_type(4)))  unsigned int v4u;
typedef __attribute__((ext_vector_type(8)))  int v8i;
typedef __attribute__((ext_vector_type(4)))  int v4i;

union ABFrag { v16bf v; uint4 u[2]; };

__device__ inline __bf16 f2bf(float f) {
  union { float f; unsigned u; } in; in.f = f;
  unsigned r = (in.u + 0x7fffu + ((in.u >> 16) & 1u)) >> 16;   // RNE
  union { unsigned short s; __bf16 b; } out; out.s = (unsigned short)r;
  return out.b;
}

// Wave-private LDS ordering: drain DScnt before cross-lane reads.
__device__ inline void wave_lds_fence() {
  __builtin_amdgcn_wave_barrier();
  asm volatile("s_wait_dscnt 0" ::: "memory");
  __builtin_amdgcn_wave_barrier();
}

__device__ inline v8f wmma_bf16(const ABFrag& a, const ABFrag& b, v8f c) {
  return __builtin_amdgcn_wmma_f32_16x16x32_bf16(
      /*neg_a=*/false, a.v, /*neg_b=*/false, b.v,
      /*c_mod=*/(short)0, c, /*reuse_a=*/false, /*reuse_b=*/false);
}

// ---------------------------------------------------------------------------
// f32 -> bf16 elementwise conversion
// ---------------------------------------------------------------------------
__global__ __launch_bounds__(256) void cvt_f32_bf16(
    const float* __restrict__ in, __bf16* __restrict__ out, int n) {
  int i = blockIdx.x * 256 + threadIdx.x;
  if (i < n) out[i] = f2bf(in[i]);
}

// ---------------------------------------------------------------------------
// Wave-tile GEMM: y[m,n] = sum_k A[m,k] * W[n,k] + bias[n]
// Block stages a 32x1024 bf16 A tile into LDS with the Tensor Data Mover,
// then each of the 8 waves computes a 32x64 output tile: per K=32 step,
// 2 A-frags (from LDS) x 4 B-frags (from global, L2-resident) = 8 WMMAs.
// mode 0: f32 out [M,N]          (final projection -> d_out)
// mode 1: bf16 out [B,H,T,Dh]    (Q)
// mode 2: bf16 out [B,H,T,Dh]    (K)
// mode 3: bf16 out [B,H,Dh,T]    (V transposed)
// ---------------------------------------------------------------------------
__global__ __launch_bounds__(256) void gemm_bf16_wmma(
    const __bf16* __restrict__ A, const __bf16* __restrict__ W,
    const float* __restrict__ bias, float* __restrict__ outF,
    __bf16* __restrict__ outB, int mode) {
  __shared__ __align__(16) __bf16 sA[32 * K_];   // 64 KiB

  const int tid   = threadIdx.x;
  const int lane  = tid & 31;
  const int l16   = lane & 15;
  const int hlf   = lane >> 4;
  const int mt32  = blockIdx.x >> 1;                  // 128 row tiles of 32
  const int nt    = (blockIdx.x & 1) * 8 + (tid >> 5); // 16 col tiles of 64

  // --- TDM: stage A[mt32*32 .. +32) x [0,1024) into LDS --------------------
  if ((tid >> 5) == 0) {
    const unsigned long long gaddr =
        (unsigned long long)(const void*)(A + (size_t)mt32 * 32 * K_);
    // Low 32 bits of a generic LDS address are the LDS byte offset (ISA 10.2).
    const unsigned lds_addr = (unsigned)(uintptr_t)(void*)sA;
    v4u g0;
    g0.x = 0x1u;                                   // count=1, user mode
    g0.y = lds_addr;                               // lds_addr (bytes)
    g0.z = (unsigned)(gaddr & 0xFFFFFFFFu);        // global_addr[31:0]
    g0.w = (unsigned)((gaddr >> 32) & 0x1FFFFFFu)  // global_addr[56:32]
           | (2u << 30);                           // type = 2 ("image")
    v8i g1;
    g1[0] = (int)(1u << 16);                       // mask=0, data_size=2B
    g1[1] = (int)((1024u & 0xFFFFu) << 16);        // tensor_dim0[15:0]=1024
    g1[2] = (int)(((1024u >> 16) & 0xFFFFu) |      // tensor_dim0[31:16]
                  ((4096u & 0xFFFFu) << 16));      // tensor_dim1[15:0]=4096
    g1[3] = (int)(((4096u >> 16) & 0xFFFFu) |      // tensor_dim1[31:16]
                  (1024u << 16));                  // tile_dim0=1024
    g1[4] = 32;                                    // tile_dim1=32, tile_dim2=0
    g1[5] = 1024;                                  // tensor_dim0_stride=1024
    g1[6] = 0;
    g1[7] = 0;
    v4i g2 = {0, 0, 0, 0};                         // 2D tensor: groups 2/3
    v4i g3 = {0, 0, 0, 0};                         //   effectively unused
    v8i g4 = {0, 0, 0, 0, 0, 0, 0, 0};             // trailing group (unused)
    __builtin_amdgcn_tensor_load_to_lds(g0, g1, g2, g3, g4, 0);
    __builtin_amdgcn_s_wait_tensorcnt(0);
  }
  // The TDM builtin carries the LDS destination as descriptor *data*, so the
  // compiler sees no write to sA and would fold its loads to undef. This asm
  // may-writes through sA, anchoring the dataflow.
  asm volatile("" :: "r"(&sA[0]) : "memory");
  __syncthreads();

  // --- compute -------------------------------------------------------------
  v8f acc[2][4];
#pragma unroll
  for (int s = 0; s < 2; ++s)
#pragma unroll
    for (int c = 0; c < 4; ++c) acc[s][c] = v8f{};

  const __bf16* sArow[2];
#pragma unroll
  for (int s = 0; s < 2; ++s)
    sArow[s] = sA + (size_t)(s * 16 + l16) * K_;
  const __bf16* wrow[4];
#pragma unroll
  for (int c = 0; c < 4; ++c)
    wrow[c] = W + (long)(nt * 64 + c * 16 + l16) * K_;

  for (int kt = 0; kt < K_; kt += 32) {
    ABFrag a[2];
#pragma unroll
    for (int s = 0; s < 2; ++s) {
      a[s].u[0] = *(const uint4*)(sArow[s] + kt + hlf * 8);
      a[s].u[1] = *(const uint4*)(sArow[s] + kt + 16 + hlf * 8);
    }
#pragma unroll
    for (int c = 0; c < 4; ++c) {
      ABFrag b;
      b.u[0] = *(const uint4*)(wrow[c] + kt + hlf * 8);
      b.u[1] = *(const uint4*)(wrow[c] + kt + 16 + hlf * 8);
#pragma unroll
      for (int s = 0; s < 2; ++s)
        acc[s][c] = wmma_bf16(a[s], b, acc[s][c]);
    }
  }

  // --- bias + scatter ------------------------------------------------------
#pragma unroll
  for (int c = 0; c < 4; ++c) {
    const int n = nt * 64 + c * 16 + l16;
    const float bv = bias[n];
#pragma unroll
    for (int s = 0; s < 2; ++s)
#pragma unroll
      for (int v = 0; v < 8; ++v) {
        const int m = mt32 * 32 + s * 16 + v + hlf * 8;
        const float val = acc[s][c][v] + bv;
        if (mode == 0) {
          outF[(long)m * N_ + n] = val;
        } else {
          const int b = m >> 11, t = m & (T_ - 1);      // T_ = 2048
          const int h = n >> 6,  d = n & (DH_ - 1);
          if (mode == 3)
            outB[(((long)(b * H_ + h)) * DH_ + d) * T_ + t] = f2bf(val);
          else
            outB[(((long)(b * H_ + h)) * T_ + t) * DH_ + d] = f2bf(val);
        }
      }
  }
}

// ---------------------------------------------------------------------------
// Flash attention: one wave per (b,h, 16-query block). Online softmax through
// per-wave LDS; all matmuls on v_wmma_f32_16x16x32_bf16.
// ---------------------------------------------------------------------------
__global__ __launch_bounds__(256) void attn_wmma(
    const __bf16* __restrict__ qb, const __bf16* __restrict__ kb,
    const __bf16* __restrict__ vtb, __bf16* __restrict__ ob) {
  __shared__ __align__(16) float  sS[8][16][32];
  __shared__ __align__(16) __bf16 sP[8][16][32];
  __shared__ float sRow[8][16];

  const int tid  = threadIdx.x;
  const int w    = tid >> 5;
  const int lane = tid & 31;
  const int l16  = lane & 15;
  const int hlf  = lane >> 4;
  const int waveId = blockIdx.x * 8 + w;   // 4096 waves = 32 bh * 128 qblocks
  const int qblk = waveId & 127;
  const int bh   = waveId >> 7;
  const int q0   = qblk * 16;
  const long baseQK = (long)bh * T_ * DH_;

  // Q fragments for d in [0,64): two A-frags, resident for the whole wave.
  ABFrag aQ[2];
  {
    const __bf16* qrow = qb + baseQK + (long)(q0 + l16) * DH_;
#pragma unroll
    for (int ds = 0; ds < 2; ++ds) {
      aQ[ds].u[0] = *(const uint4*)(qrow + ds * 32 + hlf * 8);
      aQ[ds].u[1] = *(const uint4*)(qrow + ds * 32 + 16 + hlf * 8);
    }
  }

  v8f o[4] = {v8f{}, v8f{}, v8f{}, v8f{}};
  float m_prev = -INFINITY, l_prev = 0.0f;
  const float scale = 0.125f;               // 1/sqrt(64)
  const int jbmax = (q0 + 15) >> 5;         // causal: inclusive kv-block bound

  for (int jb = 0; jb <= jbmax; ++jb) {
    const int j0 = jb * 32;

    // S = Q * K^T for a 16x32 kv chunk (2 col tiles x 2 d steps = 4 WMMAs)
    v8f s[2] = {v8f{}, v8f{}};
#pragma unroll
    for (int jn = 0; jn < 2; ++jn) {
      const __bf16* krow = kb + baseQK + (long)(j0 + jn * 16 + l16) * DH_;
#pragma unroll
      for (int ds = 0; ds < 2; ++ds) {
        ABFrag bK;
        bK.u[0] = *(const uint4*)(krow + ds * 32 + hlf * 8);
        bK.u[1] = *(const uint4*)(krow + ds * 32 + 16 + hlf * 8);
        s[jn] = wmma_bf16(aQ[ds], bK, s[jn]);
      }
    }

    // Scale + causal mask, spill C-layout tile to LDS row-major.
#pragma unroll
    for (int jn = 0; jn < 2; ++jn)
#pragma unroll
      for (int v = 0; v < 8; ++v) {
        const int m = q0 + v + hlf * 8;
        const int j = j0 + jn * 16 + l16;
        sS[w][v + hlf * 8][jn * 16 + l16] =
            (j <= m) ? s[jn][v] * scale : -10000.0f;
      }
    wave_lds_fence();

    // Online softmax: lane r (<16) owns query row r.
    if (lane < 16) {
      const int r = lane;
      float mx = -INFINITY;
#pragma unroll
      for (int c = 0; c < 32; ++c) mx = fmaxf(mx, sS[w][r][c]);
      const float mnew = fmaxf(m_prev, mx);
      const float corr = __expf(m_prev - mnew);
      float sum = 0.0f;
#pragma unroll
      for (int c = 0; c < 32; ++c) {
        const float p = __expf(sS[w][r][c] - mnew);
        sum += p;
        sP[w][r][c] = f2bf(p);
      }
      l_prev = l_prev * corr + sum;
      m_prev = mnew;
      sRow[w][r] = corr;
    }
    wave_lds_fence();

    // Rescale running output by per-row correction.
#pragma unroll
    for (int v = 0; v < 8; ++v) {
      const float cr = sRow[w][v + hlf * 8];
      o[0][v] *= cr; o[1][v] *= cr; o[2][v] *= cr; o[3][v] *= cr;
    }

    // Re-pack P into an A-fragment from LDS, then O += P * V (4 WMMAs).
    ABFrag aP;
    aP.u[0] = *(const uint4*)&sP[w][l16][hlf * 8];
    aP.u[1] = *(const uint4*)&sP[w][l16][16 + hlf * 8];
#pragma unroll
    for (int dc = 0; dc < 4; ++dc) {
      const __bf16* vrow = vtb + ((long)bh * DH_ + dc * 16 + l16) * T_ + j0;
      ABFrag bV;
      bV.u[0] = *(const uint4*)(vrow + hlf * 8);
      bV.u[1] = *(const uint4*)(vrow + 16 + hlf * 8);
      o[dc] = wmma_bf16(aP, bV, o[dc]);
    }
  }

  // Final 1/l normalization and store context as bf16 [B,T,C].
  if (lane < 16) sRow[w][lane] = 1.0f / l_prev;
  wave_lds_fence();

  const int b = bh >> 4, h = bh & 15;
#pragma unroll
  for (int v = 0; v < 8; ++v) {
    const int mloc = v + hlf * 8;
    const float inv = sRow[w][mloc];
    const long rowoff = ((long)b * T_ + q0 + mloc) * C_ + h * DH_;
#pragma unroll
    for (int dc = 0; dc < 4; ++dc)
      ob[rowoff + dc * 16 + l16] = f2bf(o[dc][v] * inv);
  }
}

// ---------------------------------------------------------------------------
// Launch: cvt -> QKV GEMMs -> attention -> output GEMM, all on `stream`.
// ---------------------------------------------------------------------------
extern "C" void kernel_launch(void* const* d_in, const int* in_sizes, int n_in,
                              void* d_out, int out_size, void* d_ws, size_t ws_size,
                              hipStream_t stream) {
  const float* x  = (const float*)d_in[0];
  const float* wq = (const float*)d_in[1];
  const float* bq = (const float*)d_in[2];
  const float* wk = (const float*)d_in[3];
  const float* bk = (const float*)d_in[4];
  const float* wv = (const float*)d_in[5];
  const float* bv = (const float*)d_in[6];
  const float* wo = (const float*)d_in[7];
  const float* bo = (const float*)d_in[8];

  uint8_t* ws = (uint8_t*)d_ws;
  const size_t MiB = (size_t)1 << 20;
  __bf16* xb  = (__bf16*)(ws);              //  8 MiB  [M,K] bf16 activations
  __bf16* wqb = (__bf16*)(ws +  8 * MiB);   //  2 MiB
  __bf16* wkb = (__bf16*)(ws + 10 * MiB);   //  2 MiB
  __bf16* wvb = (__bf16*)(ws + 12 * MiB);   //  2 MiB
  __bf16* wob = (__bf16*)(ws + 14 * MiB);   //  2 MiB
  __bf16* qbf = (__bf16*)(ws + 16 * MiB);   //  8 MiB  [B,H,T,Dh]
  __bf16* kbf = (__bf16*)(ws + 24 * MiB);   //  8 MiB  [B,H,T,Dh]
  __bf16* vtb = (__bf16*)(ws + 32 * MiB);   //  8 MiB  [B,H,Dh,T]
  __bf16* ob  = (__bf16*)(ws + 40 * MiB);   //  8 MiB  [B*T, C] context

  const int nx = M_ * K_;        // 4,194,304
  const int nw = N_ * K_;        // 1,048,576
  cvt_f32_bf16<<<(nx + 255) / 256, 256, 0, stream>>>(x,  xb,  nx);
  cvt_f32_bf16<<<(nw + 255) / 256, 256, 0, stream>>>(wq, wqb, nw);
  cvt_f32_bf16<<<(nw + 255) / 256, 256, 0, stream>>>(wk, wkb, nw);
  cvt_f32_bf16<<<(nw + 255) / 256, 256, 0, stream>>>(wv, wvb, nw);
  cvt_f32_bf16<<<(nw + 255) / 256, 256, 0, stream>>>(wo, wob, nw);

  // (M/32) row tiles x 2 blocks covering N -> 256 blocks of 8 waves
  gemm_bf16_wmma<<<256, 256, 0, stream>>>(xb, wqb, bq, nullptr, qbf, 1);
  gemm_bf16_wmma<<<256, 256, 0, stream>>>(xb, wkb, bk, nullptr, kbf, 2);
  gemm_bf16_wmma<<<256, 256, 0, stream>>>(xb, wvb, bv, nullptr, vtb, 3);

  // 32 (b,h) * 128 q-blocks = 4096 waves -> 512 blocks
  attn_wmma<<<512, 256, 0, stream>>>(qbf, kbf, vtb, ob);

  gemm_bf16_wmma<<<256, 256, 0, stream>>>(ob, wob, bo, (float*)d_out, nullptr, 0);
}
